// ChannelMoeBlock_34308198760960
// MI455X (gfx1250) — compile-verified, matrix operand order checked by reading, not run on hardware
//
#include <hip/hip_runtime.h>
#include <cstdint>
#include <cstddef>

// ---------------------------------------------------------------------------
// Types / constants
// ---------------------------------------------------------------------------
typedef __attribute__((ext_vector_type(16))) __bf16 v16bf;
typedef __attribute__((ext_vector_type(8)))  __bf16 v8bf;
typedef __attribute__((ext_vector_type(8)))  float  v8f;
typedef __attribute__((ext_vector_type(4)))  int    v4i;

constexpr int D    = 768;
constexpr int E    = 16;
constexpr int KE   = 384;     // top-k / expert hidden
constexpr int SI   = 1536;    // shared expert intermediate
constexpr int MTOK = 8 * 4096;

// GEMM tiling: 128x64 block tile, 8 waves (wave32), 32x32 wave tile, K-step 32
constexpr int BM = 128, BN = 64, KS = 32;
constexpr int LDS_STRIDE = KS + 8;  // 40 bf16 = 80 B rows; 16B-aligned sublines

#define DEV __device__ __forceinline__

// ---------------------------------------------------------------------------
// Async global->LDS copy (CDNA5 GLOBAL_LOAD_ASYNC_TO_LDS_B128), with fallback
// Verified signature: (as1 int4*, as3 int4*, imm offset, imm cpol)
// ---------------------------------------------------------------------------
#if defined(__gfx1250__) && __has_builtin(__builtin_amdgcn_global_load_async_to_lds_b128)
#define HAVE_ASYNC_LDS 1
#endif

#ifdef HAVE_ASYNC_LDS
typedef __attribute__((address_space(1))) v4i* gv4i_p;
typedef __attribute__((address_space(3))) v4i* lv4i_p;
#endif

DEV void cp_b128(const __bf16* g, __bf16* l) {
#ifdef HAVE_ASYNC_LDS
  __builtin_amdgcn_global_load_async_to_lds_b128((gv4i_p)g, (lv4i_p)l, 0, 0);
#else
  *(uint4*)l = *(const uint4*)g;
#endif
}

// Wait until at most N of this wave's async-LDS loads remain in flight.
template <int N> DEV void wait_async_le() {
#ifdef HAVE_ASYNC_LDS
#if __has_builtin(__builtin_amdgcn_s_wait_asynccnt)
  __builtin_amdgcn_s_wait_asynccnt(N);
#else
  asm volatile("s_wait_asynccnt %0" ::"i"(N) : "memory");
#endif
#endif
}

// ---------------------------------------------------------------------------
// Fast math helpers
// ---------------------------------------------------------------------------
DEV float fast_rcp(float x) {
#if __has_builtin(__builtin_amdgcn_rcpf)
  return __builtin_amdgcn_rcpf(x);   // v_rcp_f32, avoids IEEE div expansion
#else
  return 1.0f / x;
#endif
}
DEV float silu(float x) { return x * fast_rcp(1.0f + __expf(-x)); }

// ---------------------------------------------------------------------------
// WMMA helper: D = A(16x32 bf16) * B(32x16 bf16) + C(f32)
// ---------------------------------------------------------------------------
DEV v8f wmma_bf16(v16bf a, v16bf b, v8f c) {
  return __builtin_amdgcn_wmma_f32_16x16x32_bf16(
      /*neg_a=*/false, a, /*neg_b=*/false, b,
      /*c_mod=*/(short)0, c, /*reuse_a=*/false, /*reuse_b=*/false);
}

// Fragment load from row-major [rows][KS] LDS tile (stride LDS_STRIDE).
// Per ISA 7.12.2: lanes 0-15 take K 0..7 & 16..23, lanes 16-31 take
// K 8..15 & 24..31 -> exactly two ds_load_b128 per fragment.
DEV v16bf load_frag(const __bf16* S, int rowBase, int lane) {
  const int half = lane >> 4, r = lane & 15;
  const __bf16* p = S + (size_t)(rowBase + r) * LDS_STRIDE + half * 8;
  union { v16bf v; v8bf h[2]; } u;
  u.h[0] = *(const v8bf*)p;         // 16 bytes
  u.h[1] = *(const v8bf*)(p + 16);  // 16 bytes
  return u.v;
}

// ---------------------------------------------------------------------------
// Templated bf16 WMMA GEMM, double-buffered async-LDS pipeline.
//   C[M,N] = op( A[M,K] @ B0[K,N] (, A @ B1) );  BT pre-transposed [N][K].
//   SCALE : A[m,k] *= kscale[k];  DUAL : out = silu(A@B0)*(A@B1) -> bf16
//   BIAS  : += bias[n];  SILU_E : silu();  ACC : f32 out accumulates
// ---------------------------------------------------------------------------
template <bool DUAL, bool SCALE, bool BIAS, bool SILU_E, bool ACC, bool OUT_BF16>
__global__ __launch_bounds__(256)
void gemm_wmma_kernel(const __bf16* __restrict__ A, int lda,
                      const __bf16* __restrict__ BT0,
                      const __bf16* __restrict__ BT1,
                      const float* __restrict__ kscale,
                      const float* __restrict__ bias,
                      void* __restrict__ Cout, int ldc, int Ksize) {
  __shared__ alignas(16) __bf16 As [2][BM * LDS_STRIDE];
  __shared__ alignas(16) __bf16 Bs0[2][BN * LDS_STRIDE];
  __shared__ alignas(16) __bf16 Bs1[DUAL ? 2 : 1][DUAL ? BN * LDS_STRIDE : 8];

  const int tid  = threadIdx.x;
  const int lane = tid & 31;
  const int wave = tid >> 5;
  const int wm   = wave & 3;   // 4 waves along M (32 rows each)
  const int wn   = wave >> 2;  // 2 waves along N (32 cols each)
  const int m0   = blockIdx.y * BM;
  const int n0   = blockIdx.x * BN;

  // staging thread maps
  const int arow = tid >> 1, aseg = (tid & 1) * 16;  // A: 16 bf16 / thread
  const int brow = tid >> 2, bseg = (tid & 3) * 8;   // B: 8 bf16 / thread

  // per-wave async-LDS instructions issued per tile (in-order completion)
  constexpr int PER_TILE = SCALE ? (DUAL ? 2 : 1) : (DUAL ? 4 : 3);

  auto stage = [&](int buf, int k0) {
    const __bf16* asrc = A + (size_t)(m0 + arow) * lda + k0 + aseg;
    __bf16* adst = As[buf] + arow * LDS_STRIDE + aseg;
    if (SCALE) {
      v8bf a0 = *(const v8bf*)asrc;
      v8bf a1 = *(const v8bf*)(asrc + 8);
#pragma unroll
      for (int j = 0; j < 8; ++j) {
        a0[j] = (__bf16)((float)a0[j] * kscale[k0 + aseg + j]);
        a1[j] = (__bf16)((float)a1[j] * kscale[k0 + aseg + 8 + j]);
      }
      *(v8bf*)adst       = a0;
      *(v8bf*)(adst + 8) = a1;
    } else {
      cp_b128(asrc, adst);
      cp_b128(asrc + 8, adst + 8);
    }
    const size_t goff = (size_t)(n0 + brow) * Ksize + k0 + bseg;
    cp_b128(BT0 + goff, Bs0[buf] + brow * LDS_STRIDE + bseg);
    if (DUAL) cp_b128(BT1 + goff, Bs1[buf] + brow * LDS_STRIDE + bseg);
  };

  const v8f vzero = {0.f, 0.f, 0.f, 0.f, 0.f, 0.f, 0.f, 0.f};
  v8f acc[2][2], accU[2][2];
#pragma unroll
  for (int mi = 0; mi < 2; ++mi)
#pragma unroll
    for (int ni = 0; ni < 2; ++ni) { acc[mi][ni] = vzero; accU[mi][ni] = vzero; }

  stage(0, 0);                          // prologue: fill buffer 0
  int buf = 0;
  for (int k0 = 0; k0 < Ksize; k0 += KS, buf ^= 1) {
    if (k0 + KS < Ksize) {
      stage(buf ^ 1, k0 + KS);          // prefetch next tile (other buffer)
      wait_async_le<PER_TILE>();        // current tile's loads have landed
    } else {
      wait_async_le<0>();
    }
    __syncthreads();

    v16bf af[2], bfr[2];
    af[0]  = load_frag(As[buf],  wm * 32,      lane);
    af[1]  = load_frag(As[buf],  wm * 32 + 16, lane);
    bfr[0] = load_frag(Bs0[buf], wn * 32,      lane);
    bfr[1] = load_frag(Bs0[buf], wn * 32 + 16, lane);
#pragma unroll
    for (int mi = 0; mi < 2; ++mi)
#pragma unroll
      for (int ni = 0; ni < 2; ++ni)
        acc[mi][ni] = wmma_bf16(af[mi], bfr[ni], acc[mi][ni]);
    if (DUAL) {
      v16bf bu[2];
      bu[0] = load_frag(Bs1[DUAL ? buf : 0], wn * 32,      lane);
      bu[1] = load_frag(Bs1[DUAL ? buf : 0], wn * 32 + 16, lane);
#pragma unroll
      for (int mi = 0; mi < 2; ++mi)
#pragma unroll
        for (int ni = 0; ni < 2; ++ni)
          accU[mi][ni] = wmma_bf16(af[mi], bu[ni], accU[mi][ni]);
    }
    __syncthreads();
  }

  // --- epilogue: C/D layout (ISA 7.12.2): N = lane%16, M = vgpr + 8*(lane>=16)
  const int halfl = lane >> 4, nlane = lane & 15;
#pragma unroll
  for (int mi = 0; mi < 2; ++mi) {
#pragma unroll
    for (int ni = 0; ni < 2; ++ni) {
      const int col = n0 + wn * 32 + ni * 16 + nlane;
      const float bval = BIAS ? bias[col] : 0.0f;
#pragma unroll
      for (int rv = 0; rv < 8; ++rv) {
        const int row = m0 + wm * 32 + mi * 16 + halfl * 8 + rv;
        const size_t idx = (size_t)row * ldc + col;
        float v;
        if (DUAL) {
          v = silu(acc[mi][ni][rv]) * accU[mi][ni][rv];
        } else {
          v = acc[mi][ni][rv] + bval;
          if (SILU_E) v = silu(v);
        }
        if (OUT_BF16) {
          ((__bf16*)Cout)[idx] = (__bf16)v;
        } else {
          float* C = (float*)Cout;
          if (ACC) C[idx] += v; else C[idx] = v;
        }
      }
    }
  }
}

// ---------------------------------------------------------------------------
// f32 -> bf16 conversion (grid-stride), plain
// ---------------------------------------------------------------------------
__global__ void cvt_f32_bf16_kernel(const float* __restrict__ s,
                                    __bf16* __restrict__ d, size_t n) {
  size_t i = (size_t)blockIdx.x * blockDim.x + threadIdx.x;
  const size_t stride = (size_t)gridDim.x * blockDim.x;
  for (; i < n; i += stride) d[i] = (__bf16)s[i];
}

// f32 [batch][K][N] -> bf16 transposed [batch][N][K] (one-time weight prep)
__global__ void cvt_transpose_kernel(const float* __restrict__ s,
                                     __bf16* __restrict__ d,
                                     int K, int N, size_t total) {
  size_t i = (size_t)blockIdx.x * blockDim.x + threadIdx.x;
  const size_t stride = (size_t)gridDim.x * blockDim.x;
  const size_t plane = (size_t)K * N;
  for (; i < total; i += stride) {
    const size_t b = i / plane, r = i % plane;
    const size_t k = r / N, n = r % N;
    d[b * plane + n * K + k] = (__bf16)s[i];
  }
}

// ---------------------------------------------------------------------------
// Block reduction helper (256 threads)
// ---------------------------------------------------------------------------
DEV float block_reduce(float v, float* red, int tid, bool is_max) {
  red[tid] = v;
  __syncthreads();
  for (int s = 128; s > 0; s >>= 1) {
    if (tid < s) red[tid] = is_max ? fmaxf(red[tid], red[tid + s])
                                   : red[tid] + red[tid + s];
    __syncthreads();
  }
  float r = red[0];
  __syncthreads();
  return r;
}

// ---------------------------------------------------------------------------
// pe = softmax(posembed @ pos_w + pos_b)   [E, D] — tiny, plain f32
// ---------------------------------------------------------------------------
__global__ __launch_bounds__(256)
void pe_kernel(const float* __restrict__ pos, const float* __restrict__ w,
               const float* __restrict__ b, float* __restrict__ pe) {
  __shared__ float xr[D];
  __shared__ float out[D];
  __shared__ float red[256];
  const int e = blockIdx.x, tid = threadIdx.x;
  for (int k = tid; k < D; k += 256) xr[k] = pos[(size_t)e * D + k];
  __syncthreads();
  for (int n = tid; n < D; n += 256) {
    float a = b[n];
    for (int k = 0; k < D; ++k) a += xr[k] * w[(size_t)k * D + n];
    out[n] = a;
  }
  __syncthreads();
  float m = -3.0e38f;
  for (int n = tid; n < D; n += 256) m = fmaxf(m, out[n]);
  m = block_reduce(m, red, tid, true);
  float s = 0.0f;
  for (int n = tid; n < D; n += 256) {
    const float ev = __expf(out[n] - m);
    out[n] = ev;
    s += ev;
  }
  s = block_reduce(s, red, tid, false);
  const float inv = fast_rcp(s);
  for (int n = tid; n < D; n += 256) pe[(size_t)e * D + n] = out[n] * inv;
}

// ---------------------------------------------------------------------------
// Per-token: descending top-384-of-768 (ordered!), softmax over selected,
// gather x channels, scale, emit bf16 expert input row [KE].
// One 256-thread block per token; bitonic sort over 1024 padded slots.
// ---------------------------------------------------------------------------
__global__ __launch_bounds__(256)
void topk_gather_kernel(const float* __restrict__ gf,
                        const float* __restrict__ x,
                        __bf16* __restrict__ ag) {
  __shared__ float vals[1024];
  __shared__ int   idxs[1024];
  __shared__ float red[256];
  const int t = blockIdx.x, tid = threadIdx.x;
  const float* row = gf + (size_t)t * D;
  for (int i = tid; i < 1024; i += 256) {
    vals[i] = (i < D) ? row[i] : -3.0e38f;
    idxs[i] = (i < D) ? i : 0;
  }
  __syncthreads();
  for (int k = 2; k <= 1024; k <<= 1) {
    for (int j = k >> 1; j > 0; j >>= 1) {
      for (int i = tid; i < 1024; i += 256) {
        const int ixj = i ^ j;
        if (ixj > i) {
          const bool desc = ((i & k) == 0);
          const float vi = vals[i], vj = vals[ixj];
          const bool sw = desc ? (vi < vj) : (vi > vj);
          if (sw) {
            vals[i] = vj; vals[ixj] = vi;
            const int ti = idxs[i]; idxs[i] = idxs[ixj]; idxs[ixj] = ti;
          }
        }
      }
      __syncthreads();
    }
  }
  const float m = vals[0];  // sorted descending -> max first
  float loc = 0.0f;
  for (int i = tid; i < KE; i += 256) {
    const float ev = __expf(vals[i] - m);
    vals[i] = ev;
    loc += ev;
  }
  const float s = block_reduce(loc, red, tid, false);
  const float inv = fast_rcp(s);
  const float* xr = x + (size_t)t * D;
  for (int i = tid; i < KE; i += 256)
    ag[(size_t)t * KE + i] = (__bf16)(xr[idxs[i]] * vals[i] * inv);
}

// ---------------------------------------------------------------------------
// LayerNorm(eps=1e-6) per token -> bf16
// ---------------------------------------------------------------------------
__global__ __launch_bounds__(256)
void layernorm_kernel(const float* __restrict__ y, const float* __restrict__ g,
                      const float* __restrict__ b, __bf16* __restrict__ yn) {
  __shared__ float red[256];
  const int t = blockIdx.x, tid = threadIdx.x;
  const float* row = y + (size_t)t * D;
  const float v0 = row[tid], v1 = row[tid + 256], v2 = row[tid + 512];
  const float mu = block_reduce(v0 + v1 + v2, red, tid, false) * (1.0f / D);
  const float d0 = v0 - mu, d1 = v1 - mu, d2 = v2 - mu;
  const float var =
      block_reduce(d0 * d0 + d1 * d1 + d2 * d2, red, tid, false) * (1.0f / D);
  const float rs = rsqrtf(var + 1e-6f);
  __bf16* o = yn + (size_t)t * D;
  o[tid]       = (__bf16)(d0 * rs * g[tid]       + b[tid]);
  o[tid + 256] = (__bf16)(d1 * rs * g[tid + 256] + b[tid + 256]);
  o[tid + 512] = (__bf16)(d2 * rs * g[tid + 512] + b[tid + 512]);
}

// ---------------------------------------------------------------------------
// Host driver
// ---------------------------------------------------------------------------
static inline void cvt(const float* s, __bf16* d, size_t n, hipStream_t st) {
  const size_t want = (n + 255) / 256;
  const int blocks = (int)(want > 4096 ? 4096 : want);
  cvt_f32_bf16_kernel<<<blocks, 256, 0, st>>>(s, d, n);
}
static inline void cvtT(const float* s, __bf16* d, int batch, int K, int N,
                        hipStream_t st) {
  const size_t total = (size_t)batch * K * N;
  const size_t want = (total + 255) / 256;
  const int blocks = (int)(want > 4096 ? 4096 : want);
  cvt_transpose_kernel<<<blocks, 256, 0, st>>>(s, d, K, N, total);
}

extern "C" void kernel_launch(void* const* d_in, const int* in_sizes, int n_in,
                              void* d_out, int out_size, void* d_ws,
                              size_t ws_size, hipStream_t stream) {
  (void)in_sizes; (void)n_in; (void)out_size; (void)ws_size;
  const float* x      = (const float*)d_in[0];   // [MTOK, D]
  const float* posemb = (const float*)d_in[1];
  const float* pos_w  = (const float*)d_in[2];
  const float* pos_b  = (const float*)d_in[3];
  const float* gate_w = (const float*)d_in[4];
  const float* gate_b = (const float*)d_in[5];
  const float* eg_w   = (const float*)d_in[6];   // [E, KE, D]
  const float* eu_w   = (const float*)d_in[7];
  const float* ed_w   = (const float*)d_in[8];   // [E, D, D]
  const float* sg_w   = (const float*)d_in[9];   // [D, SI]
  const float* su_w   = (const float*)d_in[10];
  const float* sd_w   = (const float*)d_in[11];  // [SI, D]
  const float* ln_g   = (const float*)d_in[12];
  const float* ln_b   = (const float*)d_in[13];
  const float* m1_w   = (const float*)d_in[14];
  const float* m1_b   = (const float*)d_in[15];
  const float* m2_w   = (const float*)d_in[16];
  const float* m2_b   = (const float*)d_in[17];
  float* out = (float*)d_out;

  uint8_t* ws = (uint8_t*)d_ws;
  size_t off = 0;
  auto take = [&](size_t bytes) -> uint8_t* {
    uint8_t* p = ws + off;
    off = (off + bytes + 255) & ~(size_t)255;
    return p;
  };
  // all B-side weights stored TRANSPOSED: [N][K] bf16
  float*  pe_f = (float*) take((size_t)E * D * 4);
  __bf16* xb   = (__bf16*)take((size_t)MTOK * D * 2);
  __bf16* gwb  = (__bf16*)take((size_t)D * D * 2);        // [768][768]
  __bf16* sgb  = (__bf16*)take((size_t)D * SI * 2);       // [1536][768]
  __bf16* sub  = (__bf16*)take((size_t)D * SI * 2);       // [1536][768]
  __bf16* sdb  = (__bf16*)take((size_t)SI * D * 2);       // [768][1536]
  __bf16* egb  = (__bf16*)take((size_t)E * KE * D * 2);   // [E][768][384]
  __bf16* eub  = (__bf16*)take((size_t)E * KE * D * 2);
  __bf16* edb  = (__bf16*)take((size_t)E * D * D * 2);    // [E][768][768]
  __bf16* m1b  = (__bf16*)take((size_t)D * D * 2);
  __bf16* m2b  = (__bf16*)take((size_t)D * D * 2);
  float*  gf   = (float*) take((size_t)MTOK * D * 4);
  __bf16* ag   = (__bf16*)take((size_t)MTOK * KE * 2);
  __bf16* hbuf = (__bf16*)take((size_t)MTOK * SI * 2);
  float*  ybuf = (float*) take((size_t)MTOK * D * 4);
  __bf16* ynb  = (__bf16*)take((size_t)MTOK * D * 2);
  __bf16* t1b  = (__bf16*)take((size_t)MTOK * D * 2);

  // --- one-time bf16 conversions (weights transposed to [N][K])
  cvt (x,      xb,  (size_t)MTOK * D, stream);
  cvtT(gate_w, gwb, 1,  D,  D,  stream);
  cvtT(sg_w,   sgb, 1,  D,  SI, stream);
  cvtT(su_w,   sub, 1,  D,  SI, stream);
  cvtT(sd_w,   sdb, 1,  SI, D,  stream);
  cvtT(eg_w,   egb, E,  KE, D,  stream);
  cvtT(eu_w,   eub, E,  KE, D,  stream);
  cvtT(ed_w,   edb, E,  D,  D,  stream);
  cvtT(m1_w,   m1b, 1,  D,  D,  stream);
  cvtT(m2_w,   m2b, 1,  D,  D,  stream);

  // --- pe = softmax(posembed @ pos_w + pos_b)
  pe_kernel<<<E, 256, 0, stream>>>(posemb, pos_w, pos_b, pe_f);

  const dim3 blk(256);
  const dim3 gN768 (D / BN,  MTOK / BM);   // (12, 256)
  const dim3 gN1536(SI / BN, MTOK / BM);   // (24, 256)

  // --- shared expert: h = silu(x@sg)*(x@su); y = h@sd
  gemm_wmma_kernel<true,  false, false, false, false, true>
      <<<gN1536, blk, 0, stream>>>(xb, D, sgb, sub, nullptr, nullptr,
                                   hbuf, SI, D);
  gemm_wmma_kernel<false, false, false, false, false, false>
      <<<gN768, blk, 0, stream>>>(hbuf, SI, sdb, nullptr, nullptr, nullptr,
                                  ybuf, D, SI);

  // --- 16 routed channel experts
  for (int e = 0; e < E; ++e) {
    gemm_wmma_kernel<false, true, true, false, false, false>
        <<<gN768, blk, 0, stream>>>(xb, D, gwb, nullptr,
                                    pe_f + (size_t)e * D, gate_b, gf, D, D);
    topk_gather_kernel<<<MTOK, 256, 0, stream>>>(gf, x, ag);
    gemm_wmma_kernel<true,  false, false, false, false, true>
        <<<gN768, blk, 0, stream>>>(ag, KE, egb + (size_t)e * KE * D,
                                    eub + (size_t)e * KE * D, nullptr,
                                    nullptr, hbuf, D, KE);
    gemm_wmma_kernel<false, false, false, false, true, false>
        <<<gN768, blk, 0, stream>>>(hbuf, D, edb + (size_t)e * D * D, nullptr,
                                    nullptr, nullptr, ybuf, D, D);
  }

  // --- LayerNorm -> bf16
  layernorm_kernel<<<MTOK, 256, 0, stream>>>(ybuf, ln_g, ln_b, ynb);

  // --- final MLP: out = silu(yn@m1+b1) @ m2 + b2
  gemm_wmma_kernel<false, false, true, true, false, true>
      <<<gN768, blk, 0, stream>>>(ynb, D, m1b, nullptr, nullptr, m1_b,
                                  t1b, D, D);
  gemm_wmma_kernel<false, false, true, false, false, false>
      <<<gN768, blk, 0, stream>>>(t1b, D, m2b, nullptr, nullptr, m2_b,
                                  out, D, D);
}
